// SimplicialAttention_9509057593422
// MI455X (gfx1250) — matrix operand order repaired and checked by reading, not compiled, not run on hardware
//
#include <hip/hip_runtime.h>
#include <hip/hip_bf16.h>

typedef __bf16 bf16_t;
typedef __attribute__((ext_vector_type(16))) __bf16 v16bf;
typedef __attribute__((ext_vector_type(8)))  float  v8f;

#define S_LEN 1024
#define NH    4
#define DH    64
#define HD    256      // NH*DH
#define W1    512
#define W2    32

// LDS strides (in bf16 elements) chosen so every WMMA fragment chunk is a
// 16-byte-aligned ds_load_b128 and row strides break bank clustering.
#define K1_STR  72     // K1 window:   [512][72] bf16  (row-major j x d)
#define T_STR   72     // T = q*k2:    [32][72]  bf16
#define V1T_STR 520    // V1 transposed: [64][520] bf16 (d x j)
#define P_STR   520    // P (softmax): [32][520] bf16

// LDS carve (bytes). V1T overlays the K1 region (K1 dead after logits).
#define OFF_K1V1 0
#define SZ_K1V1  (W1 * K1_STR * 2)          // 73728 >= 64*520*2
#define OFF_T    (OFF_K1V1 + SZ_K1V1)       // 73728
#define SZ_T     (W2 * T_STR * 2)           // 4608
#define OFF_V2   (OFF_T + SZ_T)             // 78336
#define SZ_V2    (W2 * DH * 4)              // 8192
#define OFF_P    (OFF_V2 + SZ_V2)           // 86528
#define SZ_P     (W2 * P_STR * 2)           // 33280
#define OFF_U    (OFF_P + SZ_P)             // 119808
#define SZ_U     (W2 * DH * 4)              // 8192
#define OFF_RED  (OFF_U + SZ_U)             // 128000
#define SZ_RED   (256 * 4)                  // 1024
#define SMEM_BYTES (OFF_RED + SZ_RED)       // 129024

// Load a 16-element bf16 WMMA fragment as two 16-byte LDS chunks.
__device__ __forceinline__ v16bf ld_frag(const bf16_t* p0, const bf16_t* p1) {
    union { v16bf v; uint4 q[2]; } u;
    u.q[0] = *(const uint4*)p0;   // elements 0..7
    u.q[1] = *(const uint4*)p1;   // elements 8..15
    return u.v;
}

__global__ __launch_bounds__(256, 1)
void simplicial_attn_wmma(const float* __restrict__ xq,
                          const float* __restrict__ xk1,
                          const float* __restrict__ xk2,
                          const float* __restrict__ xv1,
                          const float* __restrict__ xv2,
                          float* __restrict__ out) {
    extern __shared__ char smem[];
    bf16_t* sK1  = (bf16_t*)(smem + OFF_K1V1);   // [jw][K1_STR]
    bf16_t* sV1T = (bf16_t*)(smem + OFF_K1V1);   // [d][V1T_STR]  (overlay)
    bf16_t* sT   = (bf16_t*)(smem + OFF_T);      // [a][T_STR]
    float*  sV2  = (float*) (smem + OFF_V2);     // [a][DH]
    bf16_t* sP   = (bf16_t*)(smem + OFF_P);      // [a][P_STR]
    float*  sU   = (float*) (smem + OFF_U);      // [a][DH]
    float*  sRed = (float*) (smem + OFF_RED);    // [256]

    const int i    = blockIdx.x;     // query position
    const int h    = blockIdx.y;     // head
    const int tid  = threadIdx.x;
    const int lane = tid & 31;
    const int wave = tid >> 5;       // 0..7
    const int hf   = lane >> 4;      // lane half (0/1)
    const int ln   = lane & 15;

    const int jw_min = (W1 - 1) - i; // jw valid iff jw >= jw_min  (j >= 0)
    const int a_min  = (W2 - 1) - i; // a  valid iff a  >= a_min   (idx2 >= 0)

    // ---- Phase 0: stage K1 window (bf16), T = q*k2 (bf16), V2 window (f32)
    for (int idx = tid; idx < W1 * DH; idx += 256) {
        int jw = idx >> 6, d = idx & 63;
        int j = i - (W1 - 1) + jw;
        float v = (j >= 0) ? xk1[j * HD + h * DH + d] : 0.0f;
        sK1[jw * K1_STR + d] = (bf16_t)v;
    }
    for (int idx = tid; idx < W2 * DH; idx += 256) {
        int a = idx >> 6, d = idx & 63;
        int i2 = i - (W2 - 1) + a;
        float tv = 0.0f, v2v = 0.0f;
        if (i2 >= 0) {
            tv  = xq[i * HD + h * DH + d] * xk2[i2 * HD + h * DH + d];
            v2v = xv2[i2 * HD + h * DH + d];
        }
        sT[a * T_STR + d] = (bf16_t)tv;
        sV2[a * DH + d]   = v2v;
    }
    __syncthreads();

    // ---- Phase 1: logits L[a,jw] = sum_d T[a,d]*K1[jw,d]   (32 x 512, K=64)
    // A fragments (T), reused across all j-tiles of this wave.
    v16bf afr[2][2];
    #pragma unroll
    for (int mt = 0; mt < 2; ++mt)
        #pragma unroll
        for (int kt = 0; kt < 2; ++kt) {
            const bf16_t* p = sT + (16 * mt + ln) * T_STR + 32 * kt + 8 * hf;
            afr[mt][kt] = ld_frag(p, p + 16);
        }

    v8f accL[2][4];
    #pragma unroll
    for (int ntl = 0; ntl < 4; ++ntl) {
        int jw0 = (wave * 4 + ntl) * 16;
        v8f c0 = {}; v8f c1 = {};
        #pragma unroll
        for (int kt = 0; kt < 2; ++kt) {
            const bf16_t* bp = sK1 + (jw0 + ln) * K1_STR + 32 * kt + 8 * hf;
            v16bf bfr = ld_frag(bp, bp + 16);
            c0 = __builtin_amdgcn_wmma_f32_16x16x32_bf16(false, afr[0][kt], false, bfr, (short)0, c0, false, false);
            c1 = __builtin_amdgcn_wmma_f32_16x16x32_bf16(false, afr[1][kt], false, bfr, (short)0, c1, false, false);
        }
        accL[0][ntl] = c0; accL[1][ntl] = c1;
    }

    // ---- Phase 2: scale + mask + joint softmax over all (a, jw) of this query
    const float scale = 0.125f;   // 1/sqrt(64)
    float lmax = -3.0e38f;
    #pragma unroll
    for (int mt = 0; mt < 2; ++mt)
        #pragma unroll
        for (int ntl = 0; ntl < 4; ++ntl)
            #pragma unroll
            for (int r = 0; r < 8; ++r) {
                int a  = 16 * mt + 8 * hf + r;
                int jw = (wave * 4 + ntl) * 16 + ln;
                float x = accL[mt][ntl][r] * scale;
                x = ((a >= a_min) && (jw >= jw_min)) ? x : -__builtin_inff();
                accL[mt][ntl][r] = x;
                lmax = fmaxf(lmax, x);
            }

    sRed[tid] = lmax;
    __syncthreads();
    for (int s = 128; s > 0; s >>= 1) {
        if (tid < s) sRed[tid] = fmaxf(sRed[tid], sRed[tid + s]);
        __syncthreads();
    }
    float gmax = sRed[0];
    __syncthreads();

    float lsum = 0.0f;
    #pragma unroll
    for (int mt = 0; mt < 2; ++mt)
        #pragma unroll
        for (int ntl = 0; ntl < 4; ++ntl)
            #pragma unroll
            for (int r = 0; r < 8; ++r) {
                float e = __expf(accL[mt][ntl][r] - gmax);  // masked -> 0
                accL[mt][ntl][r] = e;
                lsum += e;
            }
    sRed[tid] = lsum;
    __syncthreads();
    for (int s = 128; s > 0; s >>= 1) {
        if (tid < s) sRed[tid] += sRed[tid + s];
        __syncthreads();
    }
    float gsum = sRed[0];
    __syncthreads();

    // ---- Phase 3: write P (unnormalized, bf16); stage V1^T over K1 region
    #pragma unroll
    for (int mt = 0; mt < 2; ++mt)
        #pragma unroll
        for (int ntl = 0; ntl < 4; ++ntl)
            #pragma unroll
            for (int r = 0; r < 8; ++r) {
                int a  = 16 * mt + 8 * hf + r;
                int jw = (wave * 4 + ntl) * 16 + ln;
                sP[a * P_STR + jw] = (bf16_t)accL[mt][ntl][r];
            }
    for (int idx = tid; idx < W1 * DH; idx += 256) {
        int jw = idx >> 6, d = idx & 63;
        int j = i - (W1 - 1) + jw;
        float v = (j >= 0) ? xv1[j * HD + h * DH + d] : 0.0f;
        sV1T[d * V1T_STR + jw] = (bf16_t)v;
    }
    __syncthreads();

    // ---- Phase 4: U[a,d] = sum_j P[a,j]*V1[j,d]   (32 x 64, K=512)
    const int mt2 = wave >> 2;   // a-tile  0..1
    const int nt2 = wave & 3;    // d-tile  0..3
    v8f uacc = {};
    #pragma unroll
    for (int ks = 0; ks < 16; ++ks) {
        const bf16_t* ap = sP   + (16 * mt2 + ln) * P_STR   + 32 * ks + 8  * hf;
        const bf16_t* bp = sV1T + (16 * nt2 + ln) * V1T_STR + 32 * ks + 8  * hf;
        v16bf afu = ld_frag(ap, ap + 16);
        v16bf bfu = ld_frag(bp, bp + 16);
        uacc = __builtin_amdgcn_wmma_f32_16x16x32_bf16(false, afu, false, bfu, (short)0, uacc, false, false);
    }
    #pragma unroll
    for (int r = 0; r < 8; ++r) {
        int a = 16 * mt2 + 8 * hf + r;
        int d = 16 * nt2 + ln;
        sU[a * DH + d] = uacc[r];
    }
    __syncthreads();

    // ---- Phase 5: o[d] = (sum_a U[a,d] * V2w[a,d]) / gsum
    {
        int d = tid & 63;
        int g = tid >> 6;
        float part = 0.0f;
        #pragma unroll
        for (int k = 0; k < 8; ++k) {
            int a = g * 8 + k;
            part += sU[a * DH + d] * sV2[a * DH + d];
        }
        sRed[tid] = part;
    }
    __syncthreads();
    if (tid < 64) {
        float o = (sRed[tid] + sRed[64 + tid] + sRed[128 + tid] + sRed[192 + tid]) * (1.0f / gsum);
        out[i * HD + h * DH + tid] = o;
    }
}

extern "C" void kernel_launch(void* const* d_in, const int* in_sizes, int n_in,
                              void* d_out, int out_size, void* d_ws, size_t ws_size,
                              hipStream_t stream) {
    (void)in_sizes; (void)n_in; (void)out_size; (void)d_ws; (void)ws_size;
    const float* xq  = (const float*)d_in[0];
    const float* xk1 = (const float*)d_in[1];
    const float* xk2 = (const float*)d_in[2];
    const float* xv1 = (const float*)d_in[3];
    const float* xv2 = (const float*)d_in[4];
    float* out = (float*)d_out;

    // Allow > default dynamic LDS (gfx1250 has 320KB/WGP; we use ~126KB).
    (void)hipFuncSetAttribute((const void*)simplicial_attn_wmma,
                              hipFuncAttributeMaxDynamicSharedMemorySize,
                              SMEM_BYTES);

    dim3 grid(S_LEN, NH, 1);
    dim3 block(256, 1, 1);
    simplicial_attn_wmma<<<grid, block, SMEM_BYTES, stream>>>(xq, xk1, xk2, xv1, xv2, out);
}
